// GeoTokenizer_18476949307917
// MI455X (gfx1250) — matrix operand order repaired and checked by reading, not compiled, not run on hardware
//
#include <hip/hip_runtime.h>
#include <hip/hip_bf16.h>

typedef __attribute__((ext_vector_type(16))) __bf16 v16bf;
typedef __attribute__((ext_vector_type(8)))  __bf16 v8bf;
typedef __attribute__((ext_vector_type(8)))  float  v8f;

#define EMBED 768
#define VOCAB 4096
#define NROWS 32768
#define HIDC  256       // concatenated hidden (128 spatial + 128 feature)
#define NTILES (VOCAB / 16)
#define TILE_BYTES 24576  // 16 codebook rows x 768 bf16
#define BUF_BYTES  24832  // 16 cols x 1552B padded stride

static __device__ __forceinline__ void async_tile_b128(unsigned int lds_addr,
                                                       unsigned long long gaddr) {
  asm volatile("global_load_async_to_lds_b128 %0, %1, off"
               :: "v"(lds_addr), "v"(gaddr)
               : "memory");
}

static __device__ __forceinline__ void wait_async_le6() {
#if __has_builtin(__builtin_amdgcn_s_wait_asynccnt)
  __builtin_amdgcn_s_wait_asynccnt(6);
#else
  asm volatile("s_wait_asynccnt 0x6" ::: "memory");
#endif
}
static __device__ __forceinline__ void wait_async_zero() {
#if __has_builtin(__builtin_amdgcn_s_wait_asynccnt)
  __builtin_amdgcn_s_wait_asynccnt(0);
#else
  asm volatile("s_wait_asynccnt 0x0" ::: "memory");
#endif
}

// ---------------------------------------------------------------------------
// Kernel 1: codebook -> bf16 copy, plus c2[v] = 0.5 * ||codebook[v]||^2
// ---------------------------------------------------------------------------
__global__ void prep_codebook(const float* __restrict__ cb,
                              __bf16* __restrict__ cbb,
                              float* __restrict__ c2) {
  int row = blockIdx.x;
  int tid = threadIdx.x;
  const float* src = cb + (size_t)row * EMBED;
  __bf16* dst = cbb + (size_t)row * EMBED;
  float ssq = 0.f;
#pragma unroll
  for (int i = 0; i < 3; ++i) {
    float v = src[tid + i * 256];
    dst[tid + i * 256] = (__bf16)v;
    ssq += v * v;
  }
#pragma unroll
  for (int m = 1; m < 32; m <<= 1) ssq += __shfl_xor(ssq, m, 32);
  __shared__ float wsum[8];
  if ((tid & 31) == 0) wsum[tid >> 5] = ssq;
  __syncthreads();
  if (tid == 0) {
    float t = 0.f;
#pragma unroll
    for (int i = 0; i < 8; ++i) t += wsum[i];
    c2[row] = 0.5f * t;
  }
}

// ---------------------------------------------------------------------------
// Kernel 2: W2T[n][k] = (k<128 ? Ws2[k][n] : Wf2[k-128][n]) in bf16  (768x256)
//           b2[n] = bs2[n] + bf2[n]
// ---------------------------------------------------------------------------
__global__ void prep_w2(const float* __restrict__ Ws2, const float* __restrict__ bs2,
                        const float* __restrict__ Wf2, const float* __restrict__ bf2,
                        __bf16* __restrict__ W2T, float* __restrict__ b2) {
  int n = blockIdx.x, k = threadIdx.x;
  float v = (k < 128) ? Ws2[(size_t)k * EMBED + n]
                      : Wf2[(size_t)(k - 128) * EMBED + n];
  W2T[(size_t)n * HIDC + k] = (__bf16)v;
  if (k == 0) b2[n] = bs2[n] + bf2[n];
}

// ---------------------------------------------------------------------------
// Kernel 3: encoder. 64 rows per block (128 threads = 4 waves).
//  Phase 1 (VALU): H = [relu(coords@Ws1+bs1) | relu(feats@Wf1+bf1)] bf16 in LDS
//  Phase 2 (WMMA): combined = H @ W2T^T + b2  -> bf16 [32768][768]
// ---------------------------------------------------------------------------
__launch_bounds__(128)
__global__ void encode(const float* __restrict__ coords, const float* __restrict__ feats,
                       const float* __restrict__ Ws1, const float* __restrict__ bs1,
                       const float* __restrict__ Wf1, const float* __restrict__ bf1,
                       const __bf16* __restrict__ W2T, const float* __restrict__ b2,
                       __bf16* __restrict__ comb) {
  const int HS = 264;  // padded row stride (halves): 528B -> conflict-free ds reads
  __shared__ __align__(16) __bf16 H[64 * 264];
  __shared__ float cl[128];
  __shared__ float fl[640];
  int tid = threadIdx.x;
  int rowBase = blockIdx.x * 64;

  cl[tid] = coords[(size_t)rowBase * 2 + tid];
#pragma unroll
  for (int i = 0; i < 5; ++i)
    fl[tid + i * 128] = feats[(size_t)rowBase * 10 + tid + i * 128];
  __syncthreads();

  {  // phase 1: thread tid owns hidden column `tid` of both encoders
    float w0 = Ws1[tid], w1 = Ws1[128 + tid], b = bs1[tid];
    for (int m = 0; m < 64; ++m) {
      float h = b + cl[2 * m] * w0 + cl[2 * m + 1] * w1;
      H[m * HS + tid] = (__bf16)(h > 0.f ? h : 0.f);
    }
    float wf[10];
#pragma unroll
    for (int i = 0; i < 10; ++i) wf[i] = Wf1[i * 128 + tid];
    float bfv = bf1[tid];
    for (int m = 0; m < 64; ++m) {
      float h = bfv;
#pragma unroll
      for (int i = 0; i < 10; ++i) h += fl[m * 10 + i] * wf[i];
      H[m * HS + 128 + tid] = (__bf16)(h > 0.f ? h : 0.f);
    }
  }
  __syncthreads();

  // phase 2: each wave owns 16 rows; K=256 (8 wmma steps); N=768 (48 tiles)
  int lane = tid & 31, wave = tid >> 5;
  int lcol = lane & 15, lhi = lane >> 4;
  int mloc = wave * 16 + lcol;

  v16bf a[8];
#pragma unroll
  for (int s = 0; s < 8; ++s) {
    int k0 = s * 32 + lhi * 16;
    v8bf lo = *(const v8bf*)&H[mloc * HS + k0];
    v8bf hi = *(const v8bf*)&H[mloc * HS + k0 + 8];
    a[s] = __builtin_shufflevector(lo, hi, 0,1,2,3,4,5,6,7,8,9,10,11,12,13,14,15);
  }

  for (int nt = 0; nt < 48; ++nt) {
    int col = nt * 16 + lcol;
    v8f acc = {};
#pragma unroll
    for (int s = 0; s < 8; ++s) {
      v16bf bfr = *(const v16bf*)(W2T + (size_t)col * HIDC + s * 32 + lhi * 16);
      acc = __builtin_amdgcn_wmma_f32_16x16x32_bf16(false, a[s], false, bfr,
                                                    (short)0, acc, false, false);
    }
    float bb = b2[col];
    int rb = rowBase + wave * 16 + lhi * 8;
#pragma unroll
    for (int r = 0; r < 8; ++r)
      comb[(size_t)(rb + r) * EMBED + col] = (__bf16)(acc[r] + bb);
  }
}

// ---------------------------------------------------------------------------
// Kernel 4: nearest codebook entry. 128 rows / block (256 thr = 8 waves).
//  A-panels (16 rows x K=768 per wave) register-resident.  Codebook tiles
//  double-buffered in LDS via GLOBAL_LOAD_ASYNC_TO_LDS_B128 (ASYNCcnt),
//  so the 24KB tile copy overlaps the previous tile's 24 WMMAs.
//  argmax(dot - 0.5||c||^2) == argmin distance.
// ---------------------------------------------------------------------------
__launch_bounds__(256)
__global__ void nearest(const __bf16* __restrict__ comb, const __bf16* __restrict__ cbb,
                        const float* __restrict__ c2, float* __restrict__ tokf,
                        int* __restrict__ toki) {
  const int BS = 776;  // padded per-column stride (halves): 1552B
  __shared__ __align__(16) __bf16 Bt[2][16 * 776];
  int tid = threadIdx.x, lane = tid & 31, wave = tid >> 5;
  int lcol = lane & 15, lhi = lane >> 4;
  int blockRow = blockIdx.x * 128;
  int rowA = blockRow + wave * 16 + lcol;

  const __bf16* ap = comb + (size_t)rowA * EMBED;
  v16bf a[24];
#pragma unroll
  for (int s = 0; s < 24; ++s) {
    v8bf lo = *(const v8bf*)(ap + s * 32 + lhi * 16);
    v8bf hi = *(const v8bf*)(ap + s * 32 + lhi * 16 + 8);
    a[s] = __builtin_shufflevector(lo, hi, 0,1,2,3,4,5,6,7,8,9,10,11,12,13,14,15);
  }

  // per-thread async-copy slots: 6 x 16B chunks of the 24KB tile
  unsigned int ldsOff[6];
  unsigned long long gOff[6];
  {
    unsigned int ldsBase = (unsigned int)(uintptr_t)&Bt[0][0];
#pragma unroll
    for (int j = 0; j < 6; ++j) {
      int idx = tid + j * 256;
      int c = idx / 96, o = idx % 96;
      ldsOff[j] = ldsBase + (unsigned int)(c * 1552 + o * 16);
      gOff[j] = (unsigned long long)(uintptr_t)cbb + (unsigned long long)idx * 16u;
    }
  }

  float best[8];
  int bidx[8];
#pragma unroll
  for (int r = 0; r < 8; ++r) { best[r] = -3.4e38f; bidx[r] = 0; }

  // prologue: tile 0 -> buffer 0
#pragma unroll
  for (int j = 0; j < 6; ++j) async_tile_b128(ldsOff[j], gOff[j]);

  for (int nt = 0; nt < NTILES; ++nt) {
    int cur = nt & 1, nxt = cur ^ 1;
    __syncthreads();  // all waves done reading buf[nxt] (read in the previous iteration)
    if (nt + 1 < NTILES) {
#pragma unroll
      for (int j = 0; j < 6; ++j)
        async_tile_b128(ldsOff[j] + (unsigned int)(nxt * BUF_BYTES),
                        gOff[j] + (unsigned long long)(nt + 1) * TILE_BYTES);
      wait_async_le6();   // previous batch (-> buf[cur]) has landed
    } else {
      wait_async_zero();
    }
    __syncthreads();      // every wave's portion of buf[cur] visible

    int col = nt * 16 + lcol;
    const __bf16* btc = &Bt[cur][0];
    v8f acc = {};
#pragma unroll
    for (int s = 0; s < 24; ++s) {
      const __bf16* bp = btc + lcol * BS + s * 32 + lhi * 16;
      v8bf lo = *(const v8bf*)bp;
      v8bf hi = *(const v8bf*)(bp + 8);
      v16bf bfr = __builtin_shufflevector(lo, hi, 0,1,2,3,4,5,6,7,8,9,10,11,12,13,14,15);
      acc = __builtin_amdgcn_wmma_f32_16x16x32_bf16(false, a[s], false, bfr,
                                                    (short)0, acc, false, false);
    }
    float cs = c2[col];
#pragma unroll
    for (int r = 0; r < 8; ++r) {
      float v = acc[r] - cs;
      if (v > best[r]) { best[r] = v; bidx[r] = col; }
    }
  }

  // reduce across the 16 lanes of each half-wave (same row, different columns)
#pragma unroll
  for (int r = 0; r < 8; ++r) {
#pragma unroll
    for (int m = 1; m < 16; m <<= 1) {
      float ov = __shfl_xor(best[r], m, 32);
      int oi = __shfl_xor(bidx[r], m, 32);
      if (ov > best[r] || (ov == best[r] && oi < bidx[r])) { best[r] = ov; bidx[r] = oi; }
    }
    if (lcol == 0) {
      int row = blockRow + wave * 16 + lhi * 8 + r;
      tokf[row] = (float)bidx[r];
      toki[row] = bidx[r];
    }
  }
}

// ---------------------------------------------------------------------------
// Kernel 5: quantized = codebook[token]   (one row per block, float4 copy)
// ---------------------------------------------------------------------------
__global__ void gather(const int* __restrict__ toki, const float* __restrict__ cb,
                       float* __restrict__ outq) {
  int row = blockIdx.x;
  int tok = toki[row];
  const float4* s = (const float4*)(cb + (size_t)tok * EMBED);
  float4* d = (float4*)(outq + (size_t)row * EMBED);
  d[threadIdx.x] = s[threadIdx.x];
}

// ---------------------------------------------------------------------------
extern "C" void kernel_launch(void* const* d_in, const int* in_sizes, int n_in,
                              void* d_out, int out_size, void* d_ws, size_t ws_size,
                              hipStream_t stream) {
  (void)in_sizes; (void)n_in; (void)out_size; (void)ws_size;
  const float* coords = (const float*)d_in[0];
  const float* feats  = (const float*)d_in[1];
  const float* Ws1 = (const float*)d_in[2];
  const float* bs1 = (const float*)d_in[3];
  const float* Ws2 = (const float*)d_in[4];
  const float* bs2 = (const float*)d_in[5];
  const float* Wf1 = (const float*)d_in[6];
  const float* bf1 = (const float*)d_in[7];
  const float* Wf2 = (const float*)d_in[8];
  const float* bf2 = (const float*)d_in[9];
  const float* cb  = (const float*)d_in[10];

  char* ws = (char*)d_ws;
  __bf16* comb = (__bf16*)(ws + 0);          // 32768*768*2  = 50,331,648
  __bf16* cbb  = (__bf16*)(ws + 50331648);   //  4096*768*2  =  6,291,456
  __bf16* W2T  = (__bf16*)(ws + 56623104);   //   768*256*2  =    393,216
  float*  c2   = (float*) (ws + 57016320);   //   4096*4     =     16,384
  float*  b2   = (float*) (ws + 57032704);   //    768*4 (pad 4096)
  int*    toki = (int*)   (ws + 57036800);   //  32768*4     =    131,072

  float* out  = (float*)d_out;
  float* tokf = out;            // first 32768 outputs: tokens
  float* outq = out + NROWS;    // then 32768*768 quantized floats

  prep_codebook<<<VOCAB, 256, 0, stream>>>(cb, cbb, c2);
  prep_w2<<<EMBED, 256, 0, stream>>>(Ws2, bs2, Wf2, bf2, W2T, b2);
  encode<<<NROWS / 64, 128, 0, stream>>>(coords, feats, Ws1, bs1, Wf1, bf1, W2T, b2, comb);
  nearest<<<NROWS / 128, 256, 0, stream>>>(comb, cbb, c2, tokf, toki);
  gather<<<NROWS, EMBED / 4, 0, stream>>>(toki, cb, outq);
}
